// FeedForward_5652176961774
// MI455X (gfx1250) — matrix operand order repaired and checked by reading, not compiled, not run on hardware
//
#include <hip/hip_runtime.h>
#include <cstdint>
#include <cstddef>

#define T_DIM 4096
#define D_DIM 1024
#define F_DIM 4096
#define E_DIM 8
#define R_DIM 16
#define LSCALE 2.0f   // lora_alpha / lora_rank

#define BM 128
#define BN 64
#define BK 32

typedef __attribute__((ext_vector_type(16))) __bf16 v16bf;
typedef __attribute__((ext_vector_type(8)))  __bf16 v8bf;
typedef __attribute__((ext_vector_type(8)))  float  v8f;
typedef __attribute__((ext_vector_type(4)))  unsigned int u32x4;
typedef __attribute__((ext_vector_type(4)))  int i32x4;

#if __has_builtin(__builtin_amdgcn_global_load_async_to_lds_b128)
#define HAS_ASYNC 1
#else
#define HAS_ASYNC 0
#endif

__device__ __forceinline__ unsigned short f2bf(float f) {
    union { float f; unsigned u; } v; v.f = f;
    unsigned u = v.u;
    u += 0x7fffu + ((u >> 16) & 1u);          // round-to-nearest-even
    return (unsigned short)(u >> 16);
}
__device__ __forceinline__ float bf2f(unsigned short h) {
    union { unsigned u; float f; } v; v.u = ((unsigned)h) << 16; return v.f;
}
__device__ __forceinline__ float wred32(float v) {
    #pragma unroll
    for (int off = 16; off > 0; off >>= 1) v += __shfl_xor(v, off, 32);
    return v;
}

// ---- global -> LDS 16-byte copy (async-to-LDS on CDNA5) ----
__device__ __forceinline__ void cp16(unsigned short* dst, const unsigned short* src) {
#if HAS_ASYNC
    __builtin_amdgcn_global_load_async_to_lds_b128(
        (__attribute__((address_space(1))) i32x4*)src,
        (__attribute__((address_space(3))) i32x4*)dst, 0, 0);
#else
    *(u32x4*)dst = *(const u32x4*)src;
#endif
}
// Wait until <= N of this wave's async loads remain outstanding (in-order completion).
template <int N>
__device__ __forceinline__ void async_wait() {
#if HAS_ASYNC
#if __has_builtin(__builtin_amdgcn_s_wait_asynccnt)
    __builtin_amdgcn_s_wait_asynccnt(N);
#else
    asm volatile("s_wait_asynccnt %0" :: "i"(N) : "memory");
#endif
#endif
}

// ---- WMMA fragment loads from LDS (row-major tiles, K-stride = BK) ----
// A (16x32 bf16): lanes 0-15 row M=lane hold K 0..7 & 16..23; lanes 16-31 hold K 8..15 & 24..31
__device__ __forceinline__ v16bf fragA(const unsigned short* s, int rowbase, int lane) {
    int r  = rowbase + (lane & 15);
    int ko = (lane & 16) ? 8 : 0;
    v8bf x0 = *(const v8bf*)(s + r * BK + ko);
    v8bf x1 = *(const v8bf*)(s + r * BK + ko + 16);
    v16bf a;
    #pragma unroll
    for (int i = 0; i < 8; ++i) { a[i] = x0[i]; a[i + 8] = x1[i]; }
    return a;
}
// B (32x16 bf16), stored as [N][K]: lane's column = lane&15, lanes<16 K 0..7 & 16..23, lanes>=16 K 8..15 & 24..31
__device__ __forceinline__ v16bf fragB(const unsigned short* s, int colbase, int lane) {
    int c  = colbase + (lane & 15);
    int ko = (lane & 16) ? 8 : 0;
    v8bf x0 = *(const v8bf*)(s + c * BK + ko);
    v8bf x1 = *(const v8bf*)(s + c * BK + ko + 16);
    v16bf b;
    #pragma unroll
    for (int i = 0; i < 8; ++i) { b[i] = x0[i]; b[i + 8] = x1[i]; }
    return b;
}
__device__ __forceinline__ v8f wmma_bf16(v16bf a, v16bf b, v8f c) {
    return __builtin_amdgcn_wmma_f32_16x16x32_bf16(false, a, false, b, (short)0, c, false, false);
}

// =================== small kernels ===================

__global__ void zero_kernel(float* __restrict__ p, int n) {
    int i = blockIdx.x * blockDim.x + threadIdx.x;
    int stride = gridDim.x * blockDim.x;
    for (; i < n; i += stride) p[i] = 0.0f;
}

__global__ void cvt_bf16_kernel(const float* __restrict__ src, unsigned short* __restrict__ dst, int n) {
    int i = blockIdx.x * blockDim.x + threadIdx.x;
    int stride = gridDim.x * blockDim.x;
    for (; i < n; i += stride) dst[i] = f2bf(src[i]);
}

// src [nrows][R] -> dst [nrows][32] zero-padded bf16
__global__ void pad_lora_b_kernel(const float* __restrict__ src, unsigned short* __restrict__ dst, int nrows) {
    int i = blockIdx.x * blockDim.x + threadIdx.x;
    int n = nrows * 32;
    int stride = gridDim.x * blockDim.x;
    for (; i < n; i += stride) {
        int c = i & 31, r = i >> 5;
        dst[i] = (c < R_DIM) ? f2bf(src[(size_t)r * R_DIM + c]) : (unsigned short)0;
    }
}

// Router: logits, softmax, top-2, renormalize. One wave per token.
__global__ __launch_bounds__(256)
void router_kernel(const float* __restrict__ x, const float* __restrict__ gw,
                   float* __restrict__ logits, float* __restrict__ route) {
    int t    = (blockIdx.x * 256 + threadIdx.x) >> 5;
    int lane = threadIdx.x & 31;
    if (t >= T_DIM) return;
    float acc[E_DIM];
    #pragma unroll
    for (int e = 0; e < E_DIM; ++e) acc[e] = 0.0f;
    for (int d = lane; d < D_DIM; d += 32) {
        float xv = x[(size_t)t * D_DIM + d];
        #pragma unroll
        for (int e = 0; e < E_DIM; ++e) acc[e] += xv * gw[(size_t)e * D_DIM + d];
    }
    #pragma unroll
    for (int e = 0; e < E_DIM; ++e) acc[e] = wred32(acc[e]);
    if (lane == 0) {
        float mx = acc[0];
        #pragma unroll
        for (int e = 1; e < E_DIM; ++e) mx = fmaxf(mx, acc[e]);
        float p[E_DIM], sum = 0.0f;
        #pragma unroll
        for (int e = 0; e < E_DIM; ++e) { p[e] = __expf(acc[e] - mx); sum += p[e]; }
        #pragma unroll
        for (int e = 0; e < E_DIM; ++e) p[e] /= sum;
        int i0 = 0;
        #pragma unroll
        for (int e = 1; e < E_DIM; ++e) if (p[e] > p[i0]) i0 = e;
        int i1 = (i0 == 0) ? 1 : 0;
        #pragma unroll
        for (int e = 0; e < E_DIM; ++e) if (e != i0 && p[e] > p[i1]) i1 = e;
        float s2 = p[i0] + p[i1];
        #pragma unroll
        for (int e = 0; e < E_DIM; ++e) {
            logits[(size_t)t * E_DIM + e] = acc[e];
            float rv = (e == i0) ? (p[i0] / s2) : ((e == i1) ? (p[i1] / s2) : 0.0f);
            route[(size_t)t * E_DIM + e] = rv;
        }
    }
}

// xA = SCALE * (x @ A[e].T), bf16, zero-padded to 32 cols. One wave per (e, token).
__global__ __launch_bounds__(256)
void lora_xa_kernel(const float* __restrict__ x, const float* __restrict__ A,
                    unsigned short* __restrict__ dst) {
    int gwv  = (blockIdx.x * 256 + threadIdx.x) >> 5;   // e * T + t
    int lane = threadIdx.x & 31;
    int e = gwv >> 12;            // T = 4096
    int t = gwv & (T_DIM - 1);
    const float* Ae = A + (size_t)e * R_DIM * D_DIM;
    float acc[R_DIM];
    #pragma unroll
    for (int r = 0; r < R_DIM; ++r) acc[r] = 0.0f;
    for (int d = lane; d < D_DIM; d += 32) {
        float xv = x[(size_t)t * D_DIM + d];
        #pragma unroll
        for (int r = 0; r < R_DIM; ++r) acc[r] += xv * Ae[(size_t)r * D_DIM + d];
    }
    #pragma unroll
    for (int r = 0; r < R_DIM; ++r) acc[r] = wred32(acc[r]);
    if (lane == 0) {
        unsigned short* o = dst + (size_t)gwv * 32;
        #pragma unroll
        for (int r = 0; r < R_DIM; ++r) o[r] = f2bf(LSCALE * acc[r]);
        #pragma unroll
        for (int r = R_DIM; r < 32; ++r) o[r] = 0;
    }
}

// hA = SCALE * (h @ A2[e].T), bf16, zero-padded. One wave per token (per-expert launch).
__global__ __launch_bounds__(256)
void lora_ha_kernel(const unsigned short* __restrict__ h, const float* __restrict__ A2e,
                    unsigned short* __restrict__ dst) {
    int t    = (blockIdx.x * 256 + threadIdx.x) >> 5;
    int lane = threadIdx.x & 31;
    if (t >= T_DIM) return;
    float acc[R_DIM];
    #pragma unroll
    for (int r = 0; r < R_DIM; ++r) acc[r] = 0.0f;
    for (int f = lane; f < F_DIM; f += 32) {
        float hv = bf2f(h[(size_t)t * F_DIM + f]);
        #pragma unroll
        for (int r = 0; r < R_DIM; ++r) acc[r] += hv * A2e[(size_t)r * F_DIM + f];
    }
    #pragma unroll
    for (int r = 0; r < R_DIM; ++r) acc[r] = wred32(acc[r]);
    if (lane == 0) {
        unsigned short* o = dst + (size_t)t * 32;
        #pragma unroll
        for (int r = 0; r < R_DIM; ++r) o[r] = f2bf(LSCALE * acc[r]);
        #pragma unroll
        for (int r = R_DIM; r < 32; ++r) o[r] = 0;
    }
}

// =================== WMMA GEMM kernels (double-buffered async LDS) ===================
// Up: G = x@W1.T + xA1e@B1e.T, U = x@W3.T + xA3e@B3e.T, h = silu(G)*U (bf16 out)
__global__ __launch_bounds__(256)
void gemm_up_kernel(const unsigned short* __restrict__ xb,
                    const unsigned short* __restrict__ w1b,
                    const unsigned short* __restrict__ w3b,
                    const unsigned short* __restrict__ xa1e,
                    const unsigned short* __restrict__ b1e,
                    const unsigned short* __restrict__ xa3e,
                    const unsigned short* __restrict__ b3e,
                    unsigned short* __restrict__ hout) {
    __shared__ __attribute__((aligned(16))) unsigned short sX [2][BM * BK];
    __shared__ __attribute__((aligned(16))) unsigned short sW1[2][BN * BK];
    __shared__ __attribute__((aligned(16))) unsigned short sW3[2][BN * BK];
    const int tid  = threadIdx.x;
    const int lane = tid & 31;
    const int wave = tid >> 5;
    const int bm = blockIdx.x * BM;
    const int bn = blockIdx.y * BN;
    const int wm = (wave & 3) * 32;
    const int wn = (wave >> 2) * 32;

    v8f accG[2][2] = {};
    v8f accU[2][2] = {};

    // 4 async loads per thread per k-tile (2 X + 1 W1 + 1 W3)
    auto issue_tiles = [&](int b, int kb) {
        #pragma unroll
        for (int it = 0; it < 2; ++it) {
            int ch = tid + it * 256;
            int r = ch >> 2, c = (ch & 3) * 8;
            cp16(&sX[b][r * BK + c], &xb[(size_t)(bm + r) * D_DIM + kb + c]);
        }
        int r = tid >> 2, c = (tid & 3) * 8;
        cp16(&sW1[b][r * BK + c], &w1b[(size_t)(bn + r) * D_DIM + kb + c]);
        cp16(&sW3[b][r * BK + c], &w3b[(size_t)(bn + r) * D_DIM + kb + c]);
    };

    issue_tiles(0, 0);
    int buf = 0;
    for (int kb = 0; kb < D_DIM; kb += BK) {
        if (kb + BK < D_DIM) {
            issue_tiles(buf ^ 1, kb + BK);   // prefetch next k-tile into other buffer
            async_wait<4>();                 // only older group must be complete
        } else {
            async_wait<0>();
        }
        __syncthreads();
        const unsigned short* cX  = sX [buf];
        const unsigned short* cW1 = sW1[buf];
        const unsigned short* cW3 = sW3[buf];
        v16bf a0  = fragA(cX,  wm,      lane);
        v16bf a1  = fragA(cX,  wm + 16, lane);
        v16bf bg0 = fragB(cW1, wn,      lane);
        v16bf bg1 = fragB(cW1, wn + 16, lane);
        v16bf bu0 = fragB(cW3, wn,      lane);
        v16bf bu1 = fragB(cW3, wn + 16, lane);
        accG[0][0] = wmma_bf16(a0, bg0, accG[0][0]);
        accG[0][1] = wmma_bf16(a0, bg1, accG[0][1]);
        accG[1][0] = wmma_bf16(a1, bg0, accG[1][0]);
        accG[1][1] = wmma_bf16(a1, bg1, accG[1][1]);
        accU[0][0] = wmma_bf16(a0, bu0, accU[0][0]);
        accU[0][1] = wmma_bf16(a0, bu1, accU[0][1]);
        accU[1][0] = wmma_bf16(a1, bu0, accU[1][0]);
        accU[1][1] = wmma_bf16(a1, bu1, accU[1][1]);
        __syncthreads();
        buf ^= 1;
    }
    // LoRA gate step (K = 32, upper 16 zero-padded; scale folded into xA)
    {
        #pragma unroll
        for (int it = 0; it < 2; ++it) {
            int ch = tid + it * 256;
            int r = ch >> 2, c = (ch & 3) * 8;
            cp16(&sX[0][r * BK + c], &xa1e[(size_t)(bm + r) * 32 + c]);
        }
        {
            int r = tid >> 2, c = (tid & 3) * 8;
            cp16(&sW1[0][r * BK + c], &b1e[(size_t)(bn + r) * 32 + c]);
        }
        async_wait<0>();
        __syncthreads();
        v16bf a0 = fragA(sX[0], wm, lane), a1 = fragA(sX[0], wm + 16, lane);
        v16bf b0 = fragB(sW1[0], wn, lane), b1v = fragB(sW1[0], wn + 16, lane);
        accG[0][0] = wmma_bf16(a0, b0,  accG[0][0]);
        accG[0][1] = wmma_bf16(a0, b1v, accG[0][1]);
        accG[1][0] = wmma_bf16(a1, b0,  accG[1][0]);
        accG[1][1] = wmma_bf16(a1, b1v, accG[1][1]);
        __syncthreads();
    }
    // LoRA up step
    {
        #pragma unroll
        for (int it = 0; it < 2; ++it) {
            int ch = tid + it * 256;
            int r = ch >> 2, c = (ch & 3) * 8;
            cp16(&sX[0][r * BK + c], &xa3e[(size_t)(bm + r) * 32 + c]);
        }
        {
            int r = tid >> 2, c = (tid & 3) * 8;
            cp16(&sW3[0][r * BK + c], &b3e[(size_t)(bn + r) * 32 + c]);
        }
        async_wait<0>();
        __syncthreads();
        v16bf a0 = fragA(sX[0], wm, lane), a1 = fragA(sX[0], wm + 16, lane);
        v16bf b0 = fragB(sW3[0], wn, lane), b1v = fragB(sW3[0], wn + 16, lane);
        accU[0][0] = wmma_bf16(a0, b0,  accU[0][0]);
        accU[0][1] = wmma_bf16(a0, b1v, accU[0][1]);
        accU[1][0] = wmma_bf16(a1, b0,  accU[1][0]);
        accU[1][1] = wmma_bf16(a1, b1v, accU[1][1]);
    }
    // Epilogue: SwiGLU -> bf16 h. C layout: vgpr i, lanes<16 -> M=i, lanes>=16 -> M=8+i; N=lane&15
    #pragma unroll
    for (int ri = 0; ri < 2; ++ri)
        #pragma unroll
        for (int ci = 0; ci < 2; ++ci)
            #pragma unroll
            for (int i = 0; i < 8; ++i) {
                float g = accG[ri][ci][i];
                float u = accU[ri][ci][i];
                float s = g / (1.0f + __expf(-g));
                int M = bm + wm + ri * 16 + ((lane >> 4) << 3) + i;
                int N = bn + wn + ci * 16 + (lane & 15);
                hout[(size_t)M * F_DIM + N] = f2bf(s * u);
            }
}

// Down: y = h@W2.T + hA2e@B2e.T ; out += route[:,e] * y
__global__ __launch_bounds__(256)
void gemm_down_kernel(const unsigned short* __restrict__ hb,
                      const unsigned short* __restrict__ w2b,
                      const unsigned short* __restrict__ ha2e,
                      const unsigned short* __restrict__ b2e,
                      const float* __restrict__ route,
                      int expert,
                      float* __restrict__ out) {
    __shared__ __attribute__((aligned(16))) unsigned short sH[2][BM * BK];
    __shared__ __attribute__((aligned(16))) unsigned short sW[2][BN * BK];
    const int tid  = threadIdx.x;
    const int lane = tid & 31;
    const int wave = tid >> 5;
    const int bm = blockIdx.x * BM;
    const int bn = blockIdx.y * BN;
    const int wm = (wave & 3) * 32;
    const int wn = (wave >> 2) * 32;

    v8f acc[2][2] = {};

    // 3 async loads per thread per k-tile (2 H + 1 W2)
    auto issue_tiles = [&](int b, int kb) {
        #pragma unroll
        for (int it = 0; it < 2; ++it) {
            int ch = tid + it * 256;
            int r = ch >> 2, c = (ch & 3) * 8;
            cp16(&sH[b][r * BK + c], &hb[(size_t)(bm + r) * F_DIM + kb + c]);
        }
        int r = tid >> 2, c = (tid & 3) * 8;
        cp16(&sW[b][r * BK + c], &w2b[(size_t)(bn + r) * F_DIM + kb + c]);
    };

    issue_tiles(0, 0);
    int buf = 0;
    for (int kb = 0; kb < F_DIM; kb += BK) {
        if (kb + BK < F_DIM) {
            issue_tiles(buf ^ 1, kb + BK);
            async_wait<3>();
        } else {
            async_wait<0>();
        }
        __syncthreads();
        const unsigned short* cH = sH[buf];
        const unsigned short* cW = sW[buf];
        v16bf a0 = fragA(cH, wm, lane), a1 = fragA(cH, wm + 16, lane);
        v16bf b0 = fragB(cW, wn, lane), b1v = fragB(cW, wn + 16, lane);
        acc[0][0] = wmma_bf16(a0, b0,  acc[0][0]);
        acc[0][1] = wmma_bf16(a0, b1v, acc[0][1]);
        acc[1][0] = wmma_bf16(a1, b0,  acc[1][0]);
        acc[1][1] = wmma_bf16(a1, b1v, acc[1][1]);
        __syncthreads();
        buf ^= 1;
    }
    // LoRA down step
    {
        #pragma unroll
        for (int it = 0; it < 2; ++it) {
            int ch = tid + it * 256;
            int r = ch >> 2, c = (ch & 3) * 8;
            cp16(&sH[0][r * BK + c], &ha2e[(size_t)(bm + r) * 32 + c]);
        }
        {
            int r = tid >> 2, c = (tid & 3) * 8;
            cp16(&sW[0][r * BK + c], &b2e[(size_t)(bn + r) * 32 + c]);
        }
        async_wait<0>();
        __syncthreads();
        v16bf a0 = fragA(sH[0], wm, lane), a1 = fragA(sH[0], wm + 16, lane);
        v16bf b0 = fragB(sW[0], wn, lane), b1v = fragB(sW[0], wn + 16, lane);
        acc[0][0] = wmma_bf16(a0, b0,  acc[0][0]);
        acc[0][1] = wmma_bf16(a0, b1v, acc[0][1]);
        acc[1][0] = wmma_bf16(a1, b0,  acc[1][0]);
        acc[1][1] = wmma_bf16(a1, b1v, acc[1][1]);
    }
    // Epilogue: weighted accumulate into out (experts are serialized launches)
    #pragma unroll
    for (int ri = 0; ri < 2; ++ri)
        #pragma unroll
        for (int ci = 0; ci < 2; ++ci)
            #pragma unroll
            for (int i = 0; i < 8; ++i) {
                int M = bm + wm + ri * 16 + ((lane >> 4) << 3) + i;
                int N = bn + wn + ci * 16 + (lane & 15);
                float w = route[(size_t)M * E_DIM + expert];
                out[(size_t)M * D_DIM + N] += w * acc[ri][ci][i];
            }
}

// =================== host launcher ===================
extern "C" void kernel_launch(void* const* d_in, const int* in_sizes, int n_in,
                              void* d_out, int out_size, void* d_ws, size_t ws_size,
                              hipStream_t stream) {
    (void)in_sizes; (void)n_in; (void)out_size; (void)ws_size;
    const float* x  = (const float*)d_in[0];
    const float* gw = (const float*)d_in[1];
    const float* W1 = (const float*)d_in[2];
    const float* W3 = (const float*)d_in[3];
    const float* W2 = (const float*)d_in[4];
    const float* A1 = (const float*)d_in[5];
    const float* B1 = (const float*)d_in[6];
    const float* A3 = (const float*)d_in[7];
    const float* B3 = (const float*)d_in[8];
    const float* A2 = (const float*)d_in[9];
    const float* B2 = (const float*)d_in[10];

    float* out    = (float*)d_out;                       // [T, D]
    float* logits = out + (size_t)T_DIM * D_DIM;         // [T, E]

    char* w = (char*)d_ws;
    auto take = [&](size_t bytes) -> char* {
        char* p = w; w += (bytes + 255) & ~(size_t)255; return p;
    };
    unsigned short* xb   = (unsigned short*)take((size_t)T_DIM * D_DIM * 2);
    unsigned short* w1b  = (unsigned short*)take((size_t)F_DIM * D_DIM * 2);
    unsigned short* w3b  = (unsigned short*)take((size_t)F_DIM * D_DIM * 2);
    unsigned short* w2b  = (unsigned short*)take((size_t)D_DIM * F_DIM * 2);
    unsigned short* b1b  = (unsigned short*)take((size_t)E_DIM * F_DIM * 32 * 2);
    unsigned short* b3b  = (unsigned short*)take((size_t)E_DIM * F_DIM * 32 * 2);
    unsigned short* b2b  = (unsigned short*)take((size_t)E_DIM * D_DIM * 32 * 2);
    unsigned short* xa1  = (unsigned short*)take((size_t)E_DIM * T_DIM * 32 * 2);
    unsigned short* xa3  = (unsigned short*)take((size_t)E_DIM * T_DIM * 32 * 2);
    unsigned short* ha2  = (unsigned short*)take((size_t)T_DIM * 32 * 2);
    unsigned short* hbuf = (unsigned short*)take((size_t)T_DIM * F_DIM * 2);
    float*          route= (float*)take((size_t)T_DIM * E_DIM * 4);

    zero_kernel<<<4096, 256, 0, stream>>>(out, T_DIM * D_DIM);
    router_kernel<<<T_DIM / 8, 256, 0, stream>>>(x, gw, logits, route);

    cvt_bf16_kernel<<<2048, 256, 0, stream>>>(x,  xb,  T_DIM * D_DIM);
    cvt_bf16_kernel<<<2048, 256, 0, stream>>>(W1, w1b, F_DIM * D_DIM);
    cvt_bf16_kernel<<<2048, 256, 0, stream>>>(W3, w3b, F_DIM * D_DIM);
    cvt_bf16_kernel<<<2048, 256, 0, stream>>>(W2, w2b, D_DIM * F_DIM);
    pad_lora_b_kernel<<<2048, 256, 0, stream>>>(B1, b1b, E_DIM * F_DIM);
    pad_lora_b_kernel<<<2048, 256, 0, stream>>>(B3, b3b, E_DIM * F_DIM);
    pad_lora_b_kernel<<<512,  256, 0, stream>>>(B2, b2b, E_DIM * D_DIM);

    lora_xa_kernel<<<E_DIM * T_DIM / 8, 256, 0, stream>>>(x, A1, xa1);
    lora_xa_kernel<<<E_DIM * T_DIM / 8, 256, 0, stream>>>(x, A3, xa3);

    dim3 gUp(T_DIM / BM, F_DIM / BN);   // 32 x 64
    dim3 gDn(T_DIM / BM, D_DIM / BN);   // 32 x 16
    for (int e = 0; e < E_DIM; ++e) {
        gemm_up_kernel<<<gUp, 256, 0, stream>>>(
            xb, w1b, w3b,
            xa1 + (size_t)e * T_DIM * 32, b1b + (size_t)e * F_DIM * 32,
            xa3 + (size_t)e * T_DIM * 32, b3b + (size_t)e * F_DIM * 32,
            hbuf);
        lora_ha_kernel<<<T_DIM / 8, 256, 0, stream>>>(hbuf, A2 + (size_t)e * R_DIM * F_DIM, ha2);
        gemm_down_kernel<<<gDn, 256, 0, stream>>>(
            hbuf, w2b, ha2, b2b + (size_t)e * D_DIM * 32, route, e, out);
    }
}
